// TransformerLM_3762391351801
// MI455X (gfx1250) — matrix operand order repaired and checked by reading, not compiled, not run on hardware
//
#include <hip/hip_runtime.h>
#include <hip/hip_bf16.h>
#include <cstdint>

typedef __bf16 bf16;
typedef __attribute__((ext_vector_type(16))) __bf16 v16bf;
typedef __attribute__((ext_vector_type(8)))  __bf16 v8bf;
typedef __attribute__((ext_vector_type(8)))  float  v8f;

#define DMODEL 1024
#define NHEAD  16
#define HDIM   64
#define DFFN   4096
#define NLAYER 4
#define SEQ    2048
#define BATCH  2
#define MROWS  (BATCH*SEQ)      // 4096
#define VOCAB  32000
#define QKVS   (3*DMODEL)       // 3072

// ---------------------------------------------------------------------------
// CDNA5 async global->LDS copy (ASYNCcnt path), with safe fallback
// ---------------------------------------------------------------------------
#if __has_builtin(__builtin_amdgcn_global_load_async_to_lds_b128)
#define HAVE_ASYNC_LDS 1
#else
#define HAVE_ASYNC_LDS 0
#endif

typedef int i32x4 __attribute__((vector_size(16)));
typedef __attribute__((address_space(1))) i32x4* as1_i32x4_ptr;
typedef __attribute__((address_space(3))) i32x4* as3_i32x4_ptr;

__device__ __forceinline__ void async_copy_b128(const bf16* g, __bf16* l) {
#if HAVE_ASYNC_LDS
  // per-lane: LDS[laddr] = MEM[gaddr], 16 bytes, tracked by ASYNCcnt
  __builtin_amdgcn_global_load_async_to_lds_b128(
      (as1_i32x4_ptr)(uintptr_t)g,
      (as3_i32x4_ptr)(uint32_t)(uintptr_t)l,
      0, 0);
#else
  *(v8bf*)l = *(const v8bf*)g;
#endif
}

__device__ __forceinline__ void async_wait_all() {
#if HAVE_ASYNC_LDS
#if __has_builtin(__builtin_amdgcn_s_wait_asynccnt)
  __builtin_amdgcn_s_wait_asynccnt(0);
#else
  asm volatile("s_wait_asynccnt 0x0" ::: "memory");
#endif
#endif
}

// ---------------------------------------------------------------------------
// helpers
// ---------------------------------------------------------------------------
__device__ inline v8f wmma_bf16(v16bf a, v16bf b, v8f c) {
  return __builtin_amdgcn_wmma_f32_16x16x32_bf16(false, a, false, b, (short)0, c,
                                                 false, false);
}

__device__ inline v16bf join8(v8bf lo, v8bf hi) {
  return __builtin_shufflevector(lo, hi, 0,1,2,3,4,5,6,7,8,9,10,11,12,13,14,15);
}

__device__ inline v8f v8f_zero() {
  v8f z;
  #pragma unroll
  for (int i = 0; i < 8; ++i) z[i] = 0.0f;
  return z;
}

__device__ inline float gelu_exact(float x) {
  return 0.5f * x * (1.0f + erff(x * 0.70710678118654752f));
}

// ---------------------------------------------------------------------------
// one-time (per launch) weight preparation
// ---------------------------------------------------------------------------
__global__ __launch_bounds__(256) void convert_kernel(const float* __restrict__ in,
                                                      bf16* __restrict__ out) {
  const size_t i = ((size_t)blockIdx.x * 256 + threadIdx.x) * 4;
  const float4 f = *(const float4*)(in + i);
  out[i + 0] = (bf16)f.x; out[i + 1] = (bf16)f.y;
  out[i + 2] = (bf16)f.z; out[i + 3] = (bf16)f.w;
}

// [K][N] f32 -> [N][K] bf16 tiled transpose (32x32 tiles, 256 threads)
__global__ __launch_bounds__(256) void transpose_convert_kernel(
    const float* __restrict__ in, bf16* __restrict__ out, int K, int N) {
  __shared__ float tile[32][33];
  const int tk = blockIdx.y * 32, tn = blockIdx.x * 32;
  const int c = threadIdx.x & 31, r0 = threadIdx.x >> 5;
  #pragma unroll
  for (int rr = 0; rr < 32; rr += 8)
    tile[r0 + rr][c] = in[(size_t)(tk + r0 + rr) * N + tn + c];
  __syncthreads();
  #pragma unroll
  for (int rr = 0; rr < 32; rr += 8)
    out[(size_t)(tn + r0 + rr) * K + tk + c] = (bf16)tile[c][r0 + rr];
}

// ---------------------------------------------------------------------------
// embedding + sinusoidal positional encoding (f32 residual stream)
// ---------------------------------------------------------------------------
__global__ __launch_bounds__(256) void embed_kernel(const int* __restrict__ idx,
                                                    const float* __restrict__ tok_emb,
                                                    float* __restrict__ x) {
  const int bt  = blockIdx.x;
  const int t   = bt % SEQ;
  const int tok = idx[bt];
  const float* src = tok_emb + (size_t)tok * DMODEL;
  float* dst = x + (size_t)bt * DMODEL;
  for (int i = threadIdx.x; i < DMODEL; i += 256) {
    const int pair = i >> 1;
    const float dv = __expf((float)(2 * pair) * (-9.210340371976184f / (float)DMODEL));
    const float ang = (float)t * dv;
    const float pe  = (i & 1) ? __cosf(ang) : __sinf(ang);
    dst[i] = src[i] + pe;
  }
}

// ---------------------------------------------------------------------------
// LayerNorm: f32 in -> bf16 out. one block (8 waves) per row of D=1024
// ---------------------------------------------------------------------------
__global__ __launch_bounds__(256) void ln_kernel(const float* __restrict__ x,
                                                 const float* __restrict__ g,
                                                 const float* __restrict__ b,
                                                 bf16* __restrict__ y) {
  __shared__ float red_s[8], red_ss[8];
  const int row = blockIdx.x;
  const float* xr = x + (size_t)row * DMODEL;
  float s = 0.0f, ss = 0.0f;
  for (int i = threadIdx.x; i < DMODEL; i += 256) {
    const float v = xr[i];
    s += v; ss += v * v;
  }
  #pragma unroll
  for (int off = 16; off >= 1; off >>= 1) {
    s  += __shfl_xor(s,  off, 32);
    ss += __shfl_xor(ss, off, 32);
  }
  const int wave = threadIdx.x >> 5, lane = threadIdx.x & 31;
  if (lane == 0) { red_s[wave] = s; red_ss[wave] = ss; }
  __syncthreads();
  float ts = 0.0f, tss = 0.0f;
  #pragma unroll
  for (int w = 0; w < 8; ++w) { ts += red_s[w]; tss += red_ss[w]; }
  const float mu  = ts * (1.0f / DMODEL);
  const float var = tss * (1.0f / DMODEL) - mu * mu;
  const float inv = rsqrtf(var + 1e-5f);
  bf16* yr = y + (size_t)row * DMODEL;
  for (int i = threadIdx.x; i < DMODEL; i += 256)
    yr[i] = (bf16)((xr[i] - mu) * inv * g[i] + b[i]);
}

// ---------------------------------------------------------------------------
// all-bf16 WMMA GEMM with double-buffered async LDS staging.
//   out = epilogue( A[M,K] @ B[N,K]^T + bias, resid )
// Block tile 128x128, K-step 64; 8 waves in 4(M)x2(N); wave tile 32x64 =
// 2x4 WMMA accumulators -> 16 v_wmma per K-stage per wave.
// EPI: 0 = store f32 (head)         1 = +bias, store bf16 (qkv)
//      2 = +bias, gelu, store bf16  3 = +bias +resid, store f32 (residual x)
// ---------------------------------------------------------------------------
#define ASTR 80   // LDS row stride in bf16 elems (160 B, 32B-aligned)

__device__ __forceinline__ void stage_tiles(const bf16* __restrict__ A,
                                            const bf16* __restrict__ Bm,
                                            __bf16* as, __bf16* bs,
                                            int bm, int bn, int K, int k0, int tid) {
  #pragma unroll
  for (int it = 0; it < 4; ++it) {
    const int idx2 = it * 256 + tid;                  // 1024 v8bf's per tile
    const int r = idx2 >> 3, c8 = idx2 & 7;
    async_copy_b128(A  + (size_t)(bm + r) * K + k0 + c8 * 8, as + r * ASTR + c8 * 8);
    async_copy_b128(Bm + (size_t)(bn + r) * K + k0 + c8 * 8, bs + r * ASTR + c8 * 8);
  }
}

template <int EPI>
__global__ __launch_bounds__(256) void gemm_bf16_kernel(
    const bf16* __restrict__ A, const bf16* __restrict__ Bm,
    const float* __restrict__ bias, const float* __restrict__ resid,
    float* __restrict__ Cf, bf16* __restrict__ Cb, int M, int N, int K) {
  __shared__ __attribute__((aligned(32))) __bf16 As[2][128 * ASTR];
  __shared__ __attribute__((aligned(32))) __bf16 Bs[2][128 * ASTR];

  const int tid  = threadIdx.x;
  const int wave = tid >> 5, lane = tid & 31;
  const int lh   = lane >> 4, ln16 = lane & 15;
  const int wm   = wave & 3, wn = wave >> 2;            // 4x2 wave grid
  const int bm   = blockIdx.y * 128, bn = blockIdx.x * 128;

  v8f acc[2][4];
  #pragma unroll
  for (int i = 0; i < 2; ++i)
    #pragma unroll
    for (int j = 0; j < 4; ++j) acc[i][j] = v8f_zero();

  // prologue: fill buffer 0
  stage_tiles(A, Bm, &As[0][0], &Bs[0][0], bm, bn, K, 0, tid);
  async_wait_all();
  __syncthreads();

  for (int k0 = 0; k0 < K; k0 += 64) {
    const int cur = (k0 >> 6) & 1;
    // overlap: start async copy of next K-tile into the other buffer
    if (k0 + 64 < K)
      stage_tiles(A, Bm, &As[cur ^ 1][0], &Bs[cur ^ 1][0], bm, bn, K, k0 + 64, tid);

    // ---- compute on current buffer: 2 K-chunks of 32, 2x4 WMMA tiles ----
    #pragma unroll
    for (int kc = 0; kc < 64; kc += 32) {
      v16bf af[2], bfr[4];
      #pragma unroll
      for (int i = 0; i < 2; ++i) {
        // A-frag: lane holds 16 contiguous K = [16*lh,+16) of row (lane%16)
        const int row = wm * 32 + i * 16 + ln16;
        const v8bf lo = *(const v8bf*)&As[cur][row * ASTR + kc + 16 * lh];
        const v8bf hi = *(const v8bf*)&As[cur][row * ASTR + kc + 16 * lh + 8];
        af[i] = join8(lo, hi);
      }
      #pragma unroll
      for (int j = 0; j < 4; ++j) {
        // B-frag (same contiguous K order as A): lane holds 16 contiguous K of column n, offset 16*lh
        const int col = wn * 64 + j * 16 + ln16;
        bfr[j] = *(const v16bf*)&Bs[cur][col * ASTR + kc + lh * 16];
      }
      #pragma unroll
      for (int i = 0; i < 2; ++i)
        #pragma unroll
        for (int j = 0; j < 4; ++j)
          acc[i][j] = wmma_bf16(af[i], bfr[j], acc[i][j]);
    }

    // next buffer ready + everyone done reading current before it is reused
    async_wait_all();
    __syncthreads();
  }

  // ---- epilogue ----
  #pragma unroll
  for (int i = 0; i < 2; ++i) {
    #pragma unroll
    for (int j = 0; j < 4; ++j) {
      const int col = bn + wn * 64 + j * 16 + ln16;
      const float bv = (EPI != 0) ? bias[col] : 0.0f;
      #pragma unroll
      for (int e = 0; e < 8; ++e) {
        const int row = bm + wm * 32 + i * 16 + e + 8 * lh;
        float v = acc[i][j][e];
        if (EPI != 0) v += bv;
        if (EPI == 2) v = gelu_exact(v);
        if (EPI == 3) v += resid[(size_t)row * N + col];
        if (EPI == 0 || EPI == 3) Cf[(size_t)row * N + col] = v;
        else                      Cb[(size_t)row * N + col] = (bf16)v;
      }
    }
  }
}

// ---------------------------------------------------------------------------
// Flash attention with WMMA (causal), hd=64, all-bf16 operands, f32 softmax.
// grid: (SEQ/64, BATCH*NHEAD). block: 128 threads = 4 waves; wave owns 16 q
// rows. K chunk staged to LDS once per block (async path); V chunk staged
// transposed. Uniform key-loop bound keeps __syncthreads legal.
// ---------------------------------------------------------------------------
#define KSTR 80

__global__ __launch_bounds__(128) void attention_kernel(const bf16* __restrict__ qkv,
                                                        bf16* __restrict__ obuf) {
  __shared__ __attribute__((aligned(32))) __bf16 p_lds[4][16 * 32]; // per-wave P
  __shared__ __attribute__((aligned(32))) __bf16 k_lds[32 * KSTR];  // K [key][dim]
  __shared__ __attribute__((aligned(32))) __bf16 v_lds[64 * 32];    // V^T [dim][key]

  const int qb   = blockIdx.x;
  const int bh   = blockIdx.y;
  const int bb   = bh / NHEAD;
  const int hh   = bh % NHEAD;
  const int tid  = threadIdx.x;
  const int wave = tid >> 5, lane = tid & 31;
  const int lh   = lane >> 4, ln16 = lane & 15;

  const int q0 = qb * 64 + wave * 16;
  const bf16* qp = qkv + (size_t)bb * SEQ * QKVS + 0 * DMODEL + hh * HDIM;
  const bf16* kp = qkv + (size_t)bb * SEQ * QKVS + 1 * DMODEL + hh * HDIM;
  const bf16* vp = qkv + (size_t)bb * SEQ * QKVS + 2 * DMODEL + hh * HDIM;

  // Q fragments (A layout), direct bf16 vector loads
  v16bf qa[2];
  {
    const bf16* base = qp + (size_t)(q0 + ln16) * QKVS;
    #pragma unroll
    for (int c = 0; c < 2; ++c) {
      const v8bf lo = *(const v8bf*)(base + c * 32 + 16 * lh);
      const v8bf hi = *(const v8bf*)(base + c * 32 + 16 * lh + 8);
      qa[c] = join8(lo, hi);
    }
  }

  v8f o[4];
  #pragma unroll
  for (int t = 0; t < 4; ++t) o[t] = v8f_zero();
  float m_i[8], l_i[8];
  #pragma unroll
  for (int i = 0; i < 8; ++i) { m_i[i] = -1e30f; l_i[i] = 0.0f; }

  const int jmax = qb * 64 + 64;
  for (int j = 0; j < jmax; j += 32) {
    __syncthreads();                       // WAR on k/v/p LDS
    // stage K chunk [32 keys][64 dims], row-major == B-frag layout (async)
    #pragma unroll
    for (int t = 0; t < 2; ++t) {
      const int chunk = t * 128 + tid;     // 256 x v8bf
      const int key = chunk >> 3, c8 = chunk & 7;
      async_copy_b128(kp + (size_t)(j + key) * QKVS + c8 * 8,
                      &k_lds[key * KSTR + c8 * 8]);
    }
    // stage V chunk transposed: v_lds[dim][key]
    #pragma unroll
    for (int t = 0; t < 16; ++t) {
      const int idx2 = t * 128 + tid;      // 2048 bf16 elements
      const int key = idx2 >> 6, dim = idx2 & 63;
      v_lds[dim * 32 + key] = vp[(size_t)(j + key) * QKVS + dim];
    }
    async_wait_all();
    __syncthreads();

    // S = Q K^T for two 16-key subtiles; online softmax; P -> LDS
    #pragma unroll
    for (int kc = 0; kc < 32; kc += 16) {
      const int kbase = j + kc;
      v16bf kb[2];
      #pragma unroll
      for (int c = 0; c < 2; ++c)
        kb[c] = *(const v16bf*)&k_lds[(kc + ln16) * KSTR + c * 32 + lh * 16];
      v8f s = v8f_zero();
      s = wmma_bf16(qa[0], kb[0], s);
      s = wmma_bf16(qa[1], kb[1], s);

      const int key = kbase + ln16;
      #pragma unroll
      for (int i = 0; i < 8; ++i) {
        const int row = q0 + i + 8 * lh;
        float val = s[i] * 0.125f;          // 1/sqrt(64)
        if (key > row) val = -1e30f;        // causal mask
        float rmax = val;
        #pragma unroll
        for (int off = 8; off >= 1; off >>= 1)
          rmax = fmaxf(rmax, __shfl_xor(rmax, off, 16));
        const float mo = m_i[i];
        const float mn = fmaxf(mo, rmax);
        const float corr = __expf(mo - mn);
        const float p = __expf(val - mn);
        float ps = p;
        #pragma unroll
        for (int off = 8; off >= 1; off >>= 1)
          ps += __shfl_xor(ps, off, 16);
        m_i[i] = mn;
        l_i[i] = l_i[i] * corr + ps;
        #pragma unroll
        for (int t = 0; t < 4; ++t) o[t][i] *= corr;
        p_lds[wave][(i + 8 * lh) * 32 + kc + ln16] = (bf16)p;
      }
    }
    __syncthreads();

    // O += P(16x32) * V(32x64)
    const v8bf plo = *(const v8bf*)&p_lds[wave][ln16 * 32 + 16 * lh];
    const v8bf phi = *(const v8bf*)&p_lds[wave][ln16 * 32 + 16 * lh + 8];
    const v16bf pa = join8(plo, phi);
    #pragma unroll
    for (int t = 0; t < 4; ++t) {
      const v16bf vb = *(const v16bf*)&v_lds[(t * 16 + ln16) * 32 + lh * 16];
      o[t] = wmma_bf16(pa, vb, o[t]);
    }
  }

  #pragma unroll
  for (int t = 0; t < 4; ++t) {
    #pragma unroll
    for (int i = 0; i < 8; ++i) {
      const int row = q0 + i + 8 * lh;
      const int dim = t * 16 + ln16;
      obuf[((size_t)bb * SEQ + row) * DMODEL + hh * HDIM + dim] =
          (bf16)(o[t][i] / l_i[i]);
    }
  }
}

// ---------------------------------------------------------------------------
// host-side orchestration
// ---------------------------------------------------------------------------
extern "C" void kernel_launch(void* const* d_in, const int* in_sizes, int n_in,
                              void* d_out, int out_size, void* d_ws, size_t ws_size,
                              hipStream_t stream) {
  const int*   idx     = (const int*)  d_in[0];
  const float* tok_emb = (const float*)d_in[1];
  const float* qkv_w   = (const float*)d_in[2];
  const float* qkv_b   = (const float*)d_in[3];
  const float* out_w   = (const float*)d_in[4];
  const float* out_b   = (const float*)d_in[5];
  const float* ln1_g   = (const float*)d_in[6];
  const float* ln1_b   = (const float*)d_in[7];
  const float* ln2_g   = (const float*)d_in[8];
  const float* ln2_b   = (const float*)d_in[9];
  const float* ff_w1   = (const float*)d_in[10];
  const float* ff_b1   = (const float*)d_in[11];
  const float* ff_w2   = (const float*)d_in[12];
  const float* ff_b2   = (const float*)d_in[13];
  const float* lnf_g   = (const float*)d_in[14];
  const float* lnf_b   = (const float*)d_in[15];
  const float* head_w  = (const float*)d_in[16];

  char* ws = (char*)d_ws;
  const size_t MB = 1 << 20;
  float* x       = (float*)(ws);                  // 16 MB  f32 residual stream
  bf16*  h_bf    = (bf16*)(ws + 16  * MB);        //  8 MB  LN output
  bf16*  qkv_bf  = (bf16*)(ws + 24  * MB);        // 24 MB  qkv activations
  bf16*  obuf_bf = (bf16*)(ws + 48  * MB);        //  8 MB  attention output
  bf16*  ffh_bf  = (bf16*)(ws + 56  * MB);        // 32 MB  ffn hidden
  bf16*  wq      = (bf16*)(ws + 88  * MB);        // 24 MB  qkv_w  [3072][1024] x4
  bf16*  wo      = (bf16*)(ws + 112 * MB);        //  8 MB  out_w  [1024][1024] x4
  bf16*  w1      = (bf16*)(ws + 120 * MB);        // 32 MB  ff_w1  [4096][1024] x4
  bf16*  w2      = (bf16*)(ws + 152 * MB);        // 32 MB  ff_w2  [1024][4096] x4
  bf16*  wh      = (bf16*)(ws + 184 * MB);        // 62.5MB head_w [32000][1024]

  // ---- one-time weight prep: bf16 + [N][K] layout ----
  for (int l = 0; l < NLAYER; ++l) {
    transpose_convert_kernel<<<dim3(QKVS / 32, DMODEL / 32), 256, 0, stream>>>(
        qkv_w + (size_t)l * DMODEL * QKVS, wq + (size_t)l * QKVS * DMODEL, DMODEL, QKVS);
    transpose_convert_kernel<<<dim3(DMODEL / 32, DMODEL / 32), 256, 0, stream>>>(
        out_w + (size_t)l * DMODEL * DMODEL, wo + (size_t)l * DMODEL * DMODEL, DMODEL, DMODEL);
    transpose_convert_kernel<<<dim3(DFFN / 32, DMODEL / 32), 256, 0, stream>>>(
        ff_w1 + (size_t)l * DMODEL * DFFN, w1 + (size_t)l * DFFN * DMODEL, DMODEL, DFFN);
    transpose_convert_kernel<<<dim3(DMODEL / 32, DFFN / 32), 256, 0, stream>>>(
        ff_w2 + (size_t)l * DFFN * DMODEL, w2 + (size_t)l * DMODEL * DFFN, DFFN, DMODEL);
  }
  convert_kernel<<<(int)((size_t)VOCAB * DMODEL / 1024), 256, 0, stream>>>(head_w, wh);

  // ---- forward pass ----
  embed_kernel<<<MROWS, 256, 0, stream>>>(idx, tok_emb, x);

  for (int l = 0; l < NLAYER; ++l) {
    ln_kernel<<<MROWS, 256, 0, stream>>>(x, ln1_g + l * DMODEL, ln1_b + l * DMODEL, h_bf);
    gemm_bf16_kernel<1><<<dim3(QKVS / 128, MROWS / 128), 256, 0, stream>>>(
        h_bf, wq + (size_t)l * QKVS * DMODEL, qkv_b + (size_t)l * QKVS, nullptr,
        nullptr, qkv_bf, MROWS, QKVS, DMODEL);
    attention_kernel<<<dim3(SEQ / 64, BATCH * NHEAD), 128, 0, stream>>>(qkv_bf, obuf_bf);
    gemm_bf16_kernel<3><<<dim3(DMODEL / 128, MROWS / 128), 256, 0, stream>>>(
        obuf_bf, wo + (size_t)l * DMODEL * DMODEL, out_b + (size_t)l * DMODEL, x,
        x, nullptr, MROWS, DMODEL, DMODEL);
    ln_kernel<<<MROWS, 256, 0, stream>>>(x, ln2_g + l * DMODEL, ln2_b + l * DMODEL, h_bf);
    gemm_bf16_kernel<2><<<dim3(DFFN / 128, MROWS / 128), 256, 0, stream>>>(
        h_bf, w1 + (size_t)l * DFFN * DMODEL, ff_b1 + (size_t)l * DFFN, nullptr,
        nullptr, ffh_bf, MROWS, DFFN, DMODEL);
    gemm_bf16_kernel<3><<<dim3(DMODEL / 128, MROWS / 128), 256, 0, stream>>>(
        ffh_bf, w2 + (size_t)l * DMODEL * DFFN, ff_b2 + (size_t)l * DMODEL, x,
        x, nullptr, MROWS, DMODEL, DFFN);
  }

  ln_kernel<<<MROWS, 256, 0, stream>>>(x, lnf_g, lnf_b, h_bf);
  gemm_bf16_kernel<0><<<dim3(VOCAB / 128, MROWS / 128), 256, 0, stream>>>(
      h_bf, wh, nullptr, nullptr, (float*)d_out, nullptr, MROWS, VOCAB, DMODEL);
}
